// SSIM_68547678044733
// MI455X (gfx1250) — matrix operand order repaired and checked by reading, not compiled, not run on hardware
//
#include <hip/hip_runtime.h>
#include <hip/hip_bf16.h>

typedef __attribute__((ext_vector_type(16))) _Float16 v16h;
typedef __attribute__((ext_vector_type(2)))  _Float16 h2;
typedef __attribute__((ext_vector_type(8)))  float    v8f;
typedef __attribute__((ext_vector_type(8)))  int      v8i;

#define IMG_H 512
#define IMG_W 512
#define NIMG  32
#define TILES (NIMG * (IMG_H / 16) * (IMG_W / 16))   // 32768

// ---------------------------------------------------------------------------
// Compile-time band-matrix fragment tables.
// Gaussian ksize=11, sigma=11/6: g(d) = exp(-d^2/(2*sigma^2)) / sum, d=-5..5
// ---------------------------------------------------------------------------
constexpr double kG[11] = {
    0.0052911577, 0.0201838600, 0.0571853058, 0.1203035436, 0.1879737994,
    0.2181233685,
    0.1879737994, 0.1203035436, 0.0571853058, 0.0201838600, 0.0052911577
};

constexpr double gband(int d) {
    return (d >= -5 && d <= 5) ? kG[d + 5] : 0.0;
}

// constexpr float -> IEEE fp16 bits (round to nearest, inputs are normal or 0)
constexpr unsigned short f2h(double f) {
    if (f <= 0.0) return 0;
    int e = 0;
    double v = f;
    while (v < 1.0)  { v *= 2.0; --e; }
    while (v >= 2.0) { v *= 0.5; ++e; }
    int mi = (int)((v - 1.0) * 1024.0 + 0.5);
    if (mi == 1024) { mi = 0; ++e; }
    return (unsigned short)(((e + 15) << 10) | mi);
}

// CDNA5 16-bit A-matrix (16x32) layout: element h (0..15) of the per-lane v16h
// fragment -> K index.  Lanes 0-15: K {0..7, 16..23}; lanes 16-31: K {8..15, 24..31}.
constexpr int a_k_of_c(int lane, int h) {
    int j = h >> 1, odd = h & 1;
    int base = (j < 4) ? 0 : 16;
    return base + ((lane & 16) ? 8 : 0) + ((j & 3) << 1) + odd;
}

struct HalfTables {
    alignas(32) unsigned short bh[32][16];  // horizontal band, B-layout (32x16)
    alignas(32) unsigned short av[32][16];  // vertical band, A-layout (16x32)
};

constexpr HalfTables makeTables() {
    HalfTables t{};
    for (int l = 0; l < 32; ++l) {
        int lr = l & 15;
        // B layout: VGPR j holds rows K = (j<4?0:16) + 8*(lane/16) + 2*(j&3), +1,
        // col = lane%16.  Bh[k,c] = g[k-c-8].
        for (int j = 0; j < 8; ++j) {
            int Kb = ((j < 4) ? 0 : 16) + ((l & 16) ? 8 : 0) + 2 * (j & 3);
            t.bh[l][2 * j]     = f2h(gband(Kb - lr - 8));
            t.bh[l][2 * j + 1] = f2h(gband(Kb + 1 - lr - 8));
        }
        // A layout: Av[r,k] = g[k-r-8]
        for (int h = 0; h < 16; ++h)
            t.av[l][h] = f2h(gband(a_k_of_c(l, h) - lr - 8));
    }
    return t;
}

__constant__ HalfTables kTab = makeTables();

__device__ __forceinline__ int a_k_of(int lane, int h) { return a_k_of_c(lane, h); }

// ds_swizzle group-of-32 xor patterns: imm = (xor<<10) | (0<<5) | 0x1f
__device__ __forceinline__ int half_swap(int v) {            // lane ^= 16
    return __builtin_amdgcn_ds_swizzle(v, 0x401F);
}
template <int IMM>
__device__ __forceinline__ float swz_add(float v) {          // v + lane^k neighbor
    int i = __builtin_amdgcn_ds_swizzle(__builtin_bit_cast(int, v), IMM);
    return v + __builtin_bit_cast(float, i);
}

__global__ __launch_bounds__(256) void ssim_tile_kernel(
        const float* __restrict__ x, const float* __restrict__ y,
        float* __restrict__ partials, float* __restrict__ out_atomic, int use_ws) {
    const int lane = threadIdx.x & 31;
    const int wave = threadIdx.x >> 5;
    const int tile = blockIdx.x * 8 + wave;

    const int n  = tile >> 10;          // image
    const int tr = (tile >> 5) & 31;    // tile row
    const int tc = tile & 31;           // tile col
    const int r0 = tr << 4;
    const int c0 = tc << 4;

    const int lrow = lane & 15;
    const int khi  = (lane & 16) ? 8 : 0;
    const size_t imgBase = (size_t)n * IMG_H * IMG_W;

    // ---- Constant band fragments: two 128-bit loads each from __constant__
    const v16h bh = *reinterpret_cast<const v16h*>(&kTab.bh[lane][0]);
    const v16h av = *reinterpret_cast<const v16h*>(&kTab.av[lane][0]);

    // ---- Load raw x/y in stage-1 A-fragment layout for two 16-row strips
    // (rows r0-8..r0+7 and r0+8..r0+23), K window = cols c0-8..c0+23.
    float xv[2][16], yv[2][16];
    const bool interior = (tr >= 1) && (tr <= 30) && (tc >= 1) && (tc <= 30); // wave-uniform

    if (interior) {
        const int cb = c0 - 8 + khi;           // multiple of 8 -> 16B aligned
        const int q0 = cb >> 2;
#pragma unroll
        for (int t = 0; t < 2; ++t) {
            int r = r0 - 8 + t * 16 + lrow;
            const float4* xr = reinterpret_cast<const float4*>(x + imgBase + (size_t)r * IMG_W);
            const float4* yr = reinterpret_cast<const float4*>(y + imgBase + (size_t)r * IMG_W);
            float4 xa = xr[q0], xb = xr[q0 + 1], xc = xr[q0 + 4], xd = xr[q0 + 5];
            float4 ya = yr[q0], yb = yr[q0 + 1], yc = yr[q0 + 4], yd = yr[q0 + 5];
            xv[t][0] = xa.x; xv[t][1] = xa.y; xv[t][2]  = xa.z; xv[t][3]  = xa.w;
            xv[t][4] = xb.x; xv[t][5] = xb.y; xv[t][6]  = xb.z; xv[t][7]  = xb.w;
            xv[t][8] = xc.x; xv[t][9] = xc.y; xv[t][10] = xc.z; xv[t][11] = xc.w;
            xv[t][12] = xd.x; xv[t][13] = xd.y; xv[t][14] = xd.z; xv[t][15] = xd.w;
            yv[t][0] = ya.x; yv[t][1] = ya.y; yv[t][2]  = ya.z; yv[t][3]  = ya.w;
            yv[t][4] = yb.x; yv[t][5] = yb.y; yv[t][6]  = yb.z; yv[t][7]  = yb.w;
            yv[t][8] = yc.x; yv[t][9] = yc.y; yv[t][10] = yc.z; yv[t][11] = yc.w;
            yv[t][12] = yd.x; yv[t][13] = yd.y; yv[t][14] = yd.z; yv[t][15] = yd.w;
        }
    } else {
#pragma unroll
        for (int t = 0; t < 2; ++t) {
            int r = r0 - 8 + t * 16 + lrow;
            bool rok = (r >= 0) && (r < IMG_H);
            const float* xrow = x + imgBase + (size_t)(rok ? r : 0) * IMG_W;
            const float* yrow = y + imgBase + (size_t)(rok ? r : 0) * IMG_W;
#pragma unroll
            for (int h = 0; h < 16; ++h) {
                int col = c0 - 8 + a_k_of(lane, h);
                bool ok = rok && (col >= 0) && (col < IMG_W);
                xv[t][h] = ok ? xrow[col] : 0.0f;
                yv[t][h] = ok ? yrow[col] : 0.0f;
            }
        }
    }

    // Convert x/y to f16 once; per-map fragments are built with packed f16 ops.
    v16h xh[2], yh[2];
#pragma unroll
    for (int t = 0; t < 2; ++t)
#pragma unroll
        for (int h = 0; h < 16; ++h) {
            xh[t][h] = (_Float16)xv[t][h];
            yh[t][h] = (_Float16)yv[t][h];
        }

    // ---- Five filtered maps: x, y, x*x, y*y, x*y -> separable filter via WMMA
    const bool lowHalf = (lane & 16) == 0;
    v8f res[5];
#pragma unroll
    for (int m = 0; m < 5; ++m) {
        v16h a0, a1;
        switch (m) {
            case 0:  a0 = xh[0];         a1 = xh[1];         break;
            case 1:  a0 = yh[0];         a1 = yh[1];         break;
            case 2:  a0 = xh[0] * xh[0]; a1 = xh[1] * xh[1]; break;  // v_pk_mul_f16
            case 3:  a0 = yh[0] * yh[0]; a1 = yh[1] * yh[1]; break;
            default: a0 = xh[0] * yh[0]; a1 = xh[1] * yh[1]; break;
        }
        // Horizontal pass: two 16x16 strips of filtered rows (r0-8.. and r0+8..)
        v8f d0 = {}, d1 = {};
        d0 = __builtin_amdgcn_wmma_f32_16x16x32_f16(false, a0, false, bh, (short)0, d0, false, false);
        d1 = __builtin_amdgcn_wmma_f32_16x16x32_f16(false, a1, false, bh, (short)0, d1, false, false);

        // C/D layout -> B layout (32x16 Tmp).  Pack row pairs (2q,2q+1) into half2
        // first: then every B-VGPR is either a self-read or a lane^16 swap (ds_swizzle).
        int pk0i[4], pk1i[4];
#pragma unroll
        for (int q = 0; q < 4; ++q) {
            h2 p0 = { (_Float16)d0[2 * q], (_Float16)d0[2 * q + 1] };  // v_cvt_pk_f16_f32
            h2 p1 = { (_Float16)d1[2 * q], (_Float16)d1[2 * q + 1] };
            pk0i[q] = __builtin_bit_cast(int, p0);
            pk1i[q] = __builtin_bit_cast(int, p1);
        }
        int sw0[4], sw1[4];
#pragma unroll
        for (int q = 0; q < 4; ++q) {
            sw0[q] = half_swap(pk0i[q]);
            sw1[q] = half_swap(pk1i[q]);
        }
        v8i bti;
#pragma unroll
        for (int j = 0; j < 4; ++j) {
            bti[j]     = lowHalf ? pk0i[j] : pk0i[j];  // strip0 rows 2j,2j+1 / 2j+8,2j+9
            bti[j + 4] = lowHalf ? pk1i[j] : pk1i[j]; // strip1 rows 2j,2j+1 / 2j+8,2j+9
        }
        v16h bt = __builtin_bit_cast(v16h, bti);

        // Vertical pass
        v8f o = {};
        o = __builtin_amdgcn_wmma_f32_16x16x32_f16(false, av, false, bt, (short)0, o, false, false);
        res[m] = o;
    }

    // ---- SSIM map + in-wave reduction (8 elements / lane, C/D layout is elementwise-aligned)
    const float C1 = 0.0001f, C2 = 0.0009f;
    float partial = 0.0f;
#pragma unroll
    for (int v = 0; v < 8; ++v) {
        float mux = res[0][v], muy = res[1][v];
        float fxx = res[2][v], fyy = res[3][v], fxy = res[4][v];
        float mux2 = mux * mux, muy2 = muy * muy, muxy = mux * muy;
        float sx = fxx - mux2, sy = fyy - muy2, sxy = fxy - muxy;
        float num = (2.0f * muxy + C1) * (2.0f * sxy + C2);
        float den = (mux2 + muy2 + C1) * (sx + sy + C2);
        partial += __fdividef(num, den);       // rcp+mul, fine for SSIM ranges
    }
    partial = swz_add<0x401F>(partial);   // xor 16
    partial = swz_add<0x201F>(partial);   // xor 8
    partial = swz_add<0x101F>(partial);   // xor 4
    partial = swz_add<0x081F>(partial);   // xor 2
    partial = swz_add<0x041F>(partial);   // xor 1

    if (lane == 0) {
        if (use_ws) {
            partials[tile] = partial;
        } else {
            atomicAdd(out_atomic, partial * (1.0f / ((float)NIMG * IMG_H * IMG_W)));
        }
    }
}

__global__ __launch_bounds__(256) void ssim_reduce_kernel(
        const float* __restrict__ partials, float* __restrict__ out) {
    __shared__ float sm[256];
    float s = 0.0f;
    for (int i = threadIdx.x; i < TILES; i += 256) s += partials[i];
    sm[threadIdx.x] = s;
    __syncthreads();
    for (int step = 128; step >= 1; step >>= 1) {
        if (threadIdx.x < step) sm[threadIdx.x] += sm[threadIdx.x + step];
        __syncthreads();
    }
    if (threadIdx.x == 0)
        out[0] = sm[0] * (1.0f / ((float)NIMG * IMG_H * IMG_W));
}

extern "C" void kernel_launch(void* const* d_in, const int* in_sizes, int n_in,
                              void* d_out, int out_size, void* d_ws, size_t ws_size,
                              hipStream_t stream) {
    const float* x = (const float*)d_in[0];
    const float* y = (const float*)d_in[1];
    float* out = (float*)d_out;
    float* partials = (float*)d_ws;

    const int use_ws = (ws_size >= (size_t)TILES * sizeof(float)) ? 1 : 0;
    if (!use_ws) {
        hipMemsetAsync(out, 0, sizeof(float), stream);  // graph-capturable memset node
    }

    ssim_tile_kernel<<<TILES / 8, 256, 0, stream>>>(x, y, partials, out, use_ws);

    if (use_ws) {
        ssim_reduce_kernel<<<1, 256, 0, stream>>>(partials, out);
    }
}
